// DepthDecoder_79018808312218
// MI455X (gfx1250) — matrix operand order repaired and checked by reading, not compiled, not run on hardware
//
#include <hip/hip_runtime.h>

typedef __attribute__((ext_vector_type(16))) _Float16 v16h;
typedef __attribute__((ext_vector_type(8)))  _Float16 v8h;
typedef __attribute__((ext_vector_type(8)))  float    v8f;
typedef __attribute__((ext_vector_type(4)))  unsigned int tdm_u32x4;
typedef __attribute__((ext_vector_type(8)))  int          tdm_i32x8;
typedef __attribute__((ext_vector_type(4)))  int          tdm_i32x4;

#ifndef __has_builtin
#define __has_builtin(x) 0
#endif
#if __has_builtin(__builtin_amdgcn_tensor_load_to_lds)
#define USE_TDM 1
#if __has_include(<hip/amd_detail/amd_gfx1250_TDM.h>)
// amdgpu-toolchain (clang-23): 6-arg form
#define TDM_LOAD(g0, g1, g2, g3) \
    __builtin_amdgcn_tensor_load_to_lds(g0, g1, g2, g3, (tdm_i32x8){0,0,0,0,0,0,0,0}, 0)
#else
// ROCm 7.2 (clang-22): 5-arg form
#define TDM_LOAD(g0, g1, g2, g3) \
    __builtin_amdgcn_tensor_load_to_lds(g0, g1, g2, g3, 0)
#endif
#else
#define USE_TDM 0
#endif

__device__ __forceinline__ float elu_f(float x) {
    return x > 0.f ? x : (__expf(x) - 1.f);
}

// ===========================================================================
// Layout conversion kernels (run once per launch; bandwidth-trivial)
// ===========================================================================

// f32 NCHW -> f16 channel-tiled [B][C/16][H][W][16]
__global__ void pack_act_k(const float* __restrict__ in, _Float16* __restrict__ out,
                           size_t HW, int total)
{
    int i = blockIdx.x * blockDim.x + threadIdx.x;
    if (i >= total) return;
    const int c = i & 15;
    const size_t rest = (size_t)(i >> 4);
    const size_t p = rest % HW;
    const size_t g = rest / HW;                 // g = b*(C/16) + cg
    out[i] = (_Float16)in[(g * 16 + c) * HW + p];
}

// f32 conv weights -> f16 pre-swizzled A-fragment layout:
// dst[(((og*taps + tap)*(CIN/32) + icb)*32 + lane)*16 + e], K=half*16+e
// (contiguous K order, matching the channel-tiled B operand)
__global__ void pack_w_k(const float* __restrict__ w, _Float16* __restrict__ out,
                         int CIN, int COUT, int taps, int transposed, int total)
{
    int i = blockIdx.x * blockDim.x + threadIdx.x;
    if (i >= total) return;
    const int e = i & 15;
    const int l = (i >> 4) & 31;
    int rest = i >> 9;
    const int icb = rest % (CIN / 32);  rest /= (CIN / 32);
    const int tap = rest % taps;        rest /= taps;
    const int og  = rest;
    const int half = l >> 4, rowm = l & 15;
    const int k = half * 16 + e;
    const int oc = og * 16 + rowm;
    const int ic = icb * 32 + k;
    const float v = transposed ? w[((size_t)ic * COUT + oc) * taps + tap]
                               : w[((size_t)oc * CIN + ic) * taps + tap];
    out[i] = (_Float16)v;
}

// ===========================================================================
// 3x3 conv (pad 1) + bias + ELU: TDM-staged implicit GEMM.
// Workgroup = TW/16 waves; tile = 16 outCh x TW pixels of one output row.
// Input slab (2 ch-groups x <=3 rows x <=TW+2 px x 16ch f16) DMA'd to LDS by
// the Tensor Data Mover (double-buffered, s_wait_tensorcnt), B frags read as
// ds_load_b128, A frags are contiguous pre-swizzled global loads.
// ===========================================================================
template<int CIN, int COUT, int TW>
__global__ __launch_bounds__((TW / 16) * 32)
void conv3x3_tdm_k(const _Float16* __restrict__ act, const _Float16* __restrict__ wA,
                   const float* __restrict__ bias, _Float16* __restrict__ out,
                   int H, int W)
{
    constexpr int NICB = CIN / 32;
    __shared__ _Float16 slab[2][2 * 3 * (TW + 2) * 16];

    const int tid  = threadIdx.x;
    const int wave = tid >> 5;
    const int lane = tid & 31;
    const int half = lane >> 4;
    const int row  = lane & 15;

    int tile = blockIdx.x;
    const int tilesX = W / TW;
    const int txb = tile % tilesX;       tile /= tilesX;
    const int y   = tile % H;            tile /= H;
    const int og  = tile % (COUT >> 4);  tile /= (COUT >> 4);
    const int b   = tile;

    const size_t HW = (size_t)H * W;
    const int x0  = txb * TW;
    const int iy0 = (y > 0) ? y - 1 : 0;
    const int iy1 = (y < H - 1) ? y + 1 : H - 1;
    const int rows = iy1 - iy0 + 1;
    const int ix0 = (x0 > 0) ? x0 - 1 : 0;
    const int ix1 = (x0 + TW < W) ? x0 + TW : W - 1;
    const int width = ix1 - ix0 + 1;

#if USE_TDM
    auto issue = [&](int icb) {
        const unsigned long long ga = (unsigned long long)(size_t)
            (act + (((size_t)(b * (CIN >> 4) + icb * 2) * H + iy0) * W + ix0) * 16);
        const unsigned lds = (unsigned)(size_t)(&slab[icb & 1][0]);
        const unsigned d0 = (unsigned)width * 16;       // x extent (f16 elems)
        const unsigned d1 = (unsigned)rows;             // y extent (rows)
        const unsigned long long s0 = (unsigned long long)W * 16;   // row stride
        const unsigned long long s1 = (unsigned long long)HW * 16;  // cg stride
        tdm_u32x4 g0;
        g0[0] = 1u;                                     // count=1, flags=0
        g0[1] = lds;                                    // lds_addr
        g0[2] = (unsigned)(ga & 0xFFFFFFFFull);         // global_addr lo
        g0[3] = (unsigned)((ga >> 32) & 0x01FFFFFFull) | (2u << 30); // hi | type=2
        tdm_i32x8 g1;
        g1[0] = (int)(1u << 16);                        // data_size = 2B
        g1[1] = (int)((d0 & 0xFFFFu) << 16);            // tensor_dim0 lo16
        g1[2] = (int)((d0 >> 16) | ((d1 & 0xFFFFu) << 16));     // dim0 hi | dim1 lo
        g1[3] = (int)((d1 >> 16) | (d0 << 16));         // dim1 hi | tile_dim0
        g1[4] = (int)(d1 | (2u << 16));                 // tile_dim1 | tile_dim2=2
        g1[5] = (int)(unsigned)(s0 & 0xFFFFFFFFull);    // dim0 stride lo32
        g1[6] = (int)(((unsigned)(s0 >> 32) & 0xFFFFu) |
                      (((unsigned)(s1 & 0xFFFFull)) << 16));    // s0 hi | s1 lo16
        g1[7] = (int)(unsigned)((s1 >> 16) & 0xFFFFFFFFull);    // s1 bits 47:16
        tdm_i32x4 g2; g2[0] = 2; g2[1] = 0; g2[2] = 0; g2[3] = 0; // tensor_dim2=2
        tdm_i32x4 g3 = (tdm_i32x4){0, 0, 0, 0};
        TDM_LOAD(g0, g1, g2, g3);
    };
    if (tid < 32) issue(0);
#endif

    v8f acc = {};
    const int xw = x0 + wave * 16 + row;   // this lane's output column (N=row)
    const _Float16* wAb = wA + (size_t)og * 9 * NICB * 512 + (unsigned)lane * 16u;

    for (int icb = 0; icb < NICB; ++icb) {
#if USE_TDM
        if (tid < 32) {
            if (icb + 1 < NICB) {
                issue(icb + 1);
                __builtin_amdgcn_s_wait_tensorcnt((short)1);  // slab icb ready
            } else {
                __builtin_amdgcn_s_wait_tensorcnt((short)0);
            }
        }
        __syncthreads();
#else
        {   // cooperative global->LDS copy fallback (ds_store path)
            const int total = 2 * rows * width * 16;
            const size_t base = ((size_t)(b * (CIN >> 4) + icb * 2) * H + iy0) * W + ix0;
            for (int i = tid; i < total; i += blockDim.x) {
                const int c = i & 15;
                int t = i >> 4;
                const int px = t % width;  t /= width;
                const int r  = t % rows;
                const int z  = t / rows;
                slab[icb & 1][i] = act[(base + (size_t)z * HW + (size_t)r * W + px) * 16 + c];
            }
        }
        __syncthreads();
#endif
        const _Float16* sl = slab[icb & 1];

        #pragma unroll
        for (int ky = 0; ky < 3; ++ky) {
            const int iy  = y + ky - 1;
            const bool yok = (iy >= 0) && (iy < H);
            const int rr  = iy - iy0;
            #pragma unroll
            for (int kx = 0; kx < 3; ++kx) {
                const int tap = ky * 3 + kx;
                const int ix  = xw + kx - 1;
                const bool ok = yok && (ix >= 0) && (ix < W);
                const v16h a = *(const v16h*)(wAb + (unsigned)(tap * NICB + icb) * 512u);
                v16h bv = {};
                if (ok)
                    bv = *(const v16h*)(sl + ((unsigned)(half * rows + rr) * (unsigned)width
                                              + (unsigned)(ix - ix0)) * 16u);
                acc = __builtin_amdgcn_wmma_f32_16x16x32_f16(
                        false, a, false, bv, (short)0, acc, false, false);
            }
        }
        __syncthreads();
    }

    _Float16* ob = out + (size_t)b * (COUT >> 4) * HW * 16;
    const unsigned ooff = (((unsigned)og * H + y) * W + xw) * 16u + half * 8u;
    v8h r8;
    #pragma unroll
    for (int r = 0; r < 8; ++r)
        r8[r] = (_Float16)elu_f(acc[r] + bias[og * 16 + half * 8 + r]);
    *(v8h*)(ob + ooff) = r8;   // one contiguous b128 store per lane
}

// ===========================================================================
// ConvTranspose2d(k=3,s=2,p=1,op=1) + bias + ELU, f16 tiled in/out.
// Parity-gated gather; contiguous v16h A/B fragment loads, 32-bit offsets.
// ===========================================================================
template<int CIN, int COUT>
__global__ __launch_bounds__(32)
void convT_elu_k(const _Float16* __restrict__ act, const _Float16* __restrict__ wA,
                 const float* __restrict__ bias, _Float16* __restrict__ out,
                 int inH, int inW)
{
    constexpr int NICB = CIN / 32;
    const int H = inH * 2, W = inW * 2;
    const int lane = threadIdx.x & 31;
    const int half = lane >> 4;
    const int row  = lane & 15;

    int tile = blockIdx.x;
    const int tilesX = W >> 4;
    const int tx = tile % tilesX;        tile /= tilesX;
    const int y  = tile % H;             tile /= H;
    const int og = tile % (COUT >> 4);   tile /= (COUT >> 4);
    const int b  = tile;

    const unsigned iHW = (unsigned)(inH * inW);
    const int ox = (tx << 4) + row;
    const _Float16* actb = act + (size_t)b * (CIN >> 4) * iHW * 16;
    const _Float16* wAb  = wA + (size_t)og * 9 * NICB * 512 + (unsigned)lane * 16u;

    v8f acc = {};

    #pragma unroll
    for (int ky = 0; ky < 3; ++ky) {
        const int ty = y + 1 - ky;
        if (ty & 1) continue;                    // wave-uniform
        const int iy = ty >> 1;
        const bool yok = (iy >= 0) && (iy < inH);
        #pragma unroll
        for (int kx = 0; kx < 3; ++kx) {
            const int txv = ox + 1 - kx;         // per-lane parity gate
            const int ix  = txv >> 1;
            const bool ok = yok && ((txv & 1) == 0) && (ix >= 0) && (ix < inW);
            const unsigned tap = (unsigned)(ky * 3 + kx);
            const unsigned py  = (unsigned)((yok ? iy : 0) * inW + (ok ? ix : 0));
            for (int icb = 0; icb < NICB; ++icb) {
                if (icb + 1 < NICB)
                    __builtin_prefetch(wAb + (tap * NICB + icb + 1) * 512u, 0, 1);
                const v16h a = *(const v16h*)(wAb + (tap * NICB + (unsigned)icb) * 512u);
                v16h bv = {};
                if (ok)
                    bv = *(const v16h*)(actb +
                          ((unsigned)(icb * 2 + half) * iHW + py) * 16u);
                acc = __builtin_amdgcn_wmma_f32_16x16x32_f16(
                        false, a, false, bv, (short)0, acc, false, false);
            }
        }
    }

    _Float16* ob = out + (size_t)b * (COUT >> 4) * H * W * 16;
    const unsigned ooff = (((unsigned)og * H + y) * W + ox) * 16u + half * 8u;
    v8h r8;
    #pragma unroll
    for (int r = 0; r < 8; ++r)
        r8[r] = (_Float16)elu_f(acc[r] + bias[og * 16 + half * 8 + r]);
    *(v8h*)(ob + ooff) = r8;
}

// ===========================================================================
// 1x1 conv 32->16 over tiled-f16 input. One wave does TPW pixel-tiles (weights
// held in registers). Writes row-major q [B,16,HW] and (optionally) pixel-
// tiled q [B,HW,16] for the vectorized fused/final consumer.
// ===========================================================================
template<int TPW>
__global__ __launch_bounds__(32)
void conv1x1_k(const _Float16* __restrict__ x, const _Float16* __restrict__ wA,
               const float* __restrict__ bias, _Float16* __restrict__ qrow,
               _Float16* __restrict__ qtile, int HW_)
{
    const int lane = threadIdx.x & 31;
    const int half = lane >> 4;
    const int row  = lane & 15;
    int tile = blockIdx.x;
    const int tilesG = HW_ / (16 * TPW);
    const int tg = tile % tilesG;
    const int b  = tile / tilesG;
    const unsigned base = (unsigned)tg * 16u * TPW;

    const v16h a = *(const v16h*)(wA + (unsigned)lane * 16u);
    float bs[8];
    #pragma unroll
    for (int r = 0; r < 8; ++r) bs[r] = bias[half * 8 + r];

    const _Float16* xb = x + (size_t)b * 2 * HW_ * 16;
    _Float16* qr = qrow + (size_t)b * 16 * HW_;
    _Float16* qt = qtile ? qtile + (size_t)b * HW_ * 16 : (_Float16*)nullptr;

    #pragma unroll
    for (int t = 0; t < TPW; ++t) {
        const unsigned p = base + (unsigned)t * 16u + (unsigned)row;
        const v16h bv = *(const v16h*)(xb + ((unsigned)half * (unsigned)HW_ + p) * 16u);
        v8f acc = {};
        acc = __builtin_amdgcn_wmma_f32_16x16x32_f16(
                false, a, false, bv, (short)0, acc, false, false);
        v8h r8;
        #pragma unroll
        for (int r = 0; r < 8; ++r) {
            const float v = acc[r] + bs[r];
            r8[r] = (_Float16)v;
            qr[(unsigned)(half * 8 + r) * (unsigned)HW_ + p] = (_Float16)v;
        }
        if (qt) *(v8h*)(qt + p * 16u + half * 8u) = r8;   // b128 store
    }
}

__global__ void zero_f32_k(float* __restrict__ p, int n)
{
    int i = blockIdx.x * blockDim.x + threadIdx.x;
    if (i < n) p[i] = 0.f;
}

// attn[b][c][d] = sum_n q[c][n]*kf[d][n]; WMMA partials + f32 atomics.
__global__ __launch_bounds__(32)
void attn_accum_k(const _Float16* __restrict__ q, const _Float16* __restrict__ kf,
                  float* __restrict__ attn, int HW_, int chunks)
{
    const int lane = threadIdx.x & 31;
    const int half = lane >> 4;
    const int row  = lane & 15;
    const int chunk = blockIdx.x % chunks;
    const int b     = blockIdx.x / chunks;
    const int span  = HW_ / chunks;

    const _Float16* qb = q  + (size_t)b * 16 * HW_;
    const _Float16* kb = kf + (size_t)b * 16 * HW_;
    const unsigned qoff = (unsigned)row * (unsigned)HW_;

    v8f acc = {};
    const unsigned n_end = (unsigned)(chunk * span + span);
    for (unsigned n0 = (unsigned)(chunk * span); n0 < n_end; n0 += 32u) {
        const _Float16* qp = qb + qoff + n0;
        const v8h alo = *(const v8h*)(qp + half * 16);
        const v8h ahi = *(const v8h*)(qp + half * 16 + 8);
        v16h a;
        #pragma unroll
        for (int e = 0; e < 8; ++e) { a[e] = alo[e]; a[8 + e] = ahi[e]; }
        const v16h bv = *(const v16h*)(kb + qoff + n0 + half * 16);
        acc = __builtin_amdgcn_wmma_f32_16x16x32_f16(
                false, a, false, bv, (short)0, acc, false, false);
    }
    #pragma unroll
    for (int r = 0; r < 8; ++r)
        atomicAdd(&attn[b * 256 + (r + half * 8) * 16 + row], acc[r]);
}

__global__ void softmax16_k(float* __restrict__ attn, int rows)
{
    int r = blockIdx.x * blockDim.x + threadIdx.x;
    if (r >= rows) return;
    float* p = attn + r * 16;
    float mx = p[0];
    #pragma unroll
    for (int i = 1; i < 16; ++i) mx = fmaxf(mx, p[i]);
    float e[16], s = 0.f;
    #pragma unroll
    for (int i = 0; i < 16; ++i) { e[i] = __expf(p[i] - mx); s += e[i]; }
    const float inv = 1.f / s;
    #pragma unroll
    for (int i = 0; i < 16; ++i) p[i] = e[i] * inv;
}

// fused = attn@q + q (K padded 16->32), then 1x1 head + sigmoid -> f32 out.
// Reads pixel-tiled q: B fragment and residual are contiguous vector loads.
__global__ __launch_bounds__(32)
void fused_final_k(const _Float16* __restrict__ qtile, const float* __restrict__ attn,
                   const float* __restrict__ wfin, const float* __restrict__ bfin,
                   float* __restrict__ out, int HW_)
{
    const int lane = threadIdx.x & 31;
    const int half = lane >> 4;
    const int row  = lane & 15;
    int tile = blockIdx.x;
    const int tilesN = HW_ >> 4;
    const int tn = tile % tilesN;
    const int b  = tile / tilesN;
    const unsigned p = ((unsigned)tn << 4) + (unsigned)row;   // this lane's pixel

    const float* ab = attn + b * 256;
    v16h a;
    #pragma unroll
    for (int e = 0; e < 8; ++e) {
        a[e]     = half ? (_Float16)0.f : (_Float16)ab[row * 16 + e];      // K in [0,8); half 1 pad
        a[8 + e] = half ? (_Float16)0.f : (_Float16)ab[row * 16 + 8 + e];  // K in [8,16); half 1 pad
    }
    const _Float16* qt = qtile + (size_t)b * HW_ * 16;
    v16h bv = {};
    if (half == 0)                       // lanes 0-15 carry K=0..15; rest zero pad
        bv = *(const v16h*)(qt + p * 16u);
    v8f acc = {};
    acc = __builtin_amdgcn_wmma_f32_16x16x32_f16(
            false, a, false, bv, (short)0, acc, false, false);

    const v8h rq = *(const v8h*)(qt + p * 16u + half * 8u);   // residual slice
    float partial = 0.f;
    #pragma unroll
    for (int r = 0; r < 8; ++r)
        partial += wfin[r + half * 8] * (acc[r] + (float)rq[r]);
    partial += __shfl_xor(partial, 16, 32);
    if (half == 0)
        out[(size_t)b * HW_ + p] = 1.f / (1.f + __expf(-(partial + bfin[0])));
}

// ===========================================================================
extern "C" void kernel_launch(void* const* d_in, const int* in_sizes, int n_in,
                              void* d_out, int out_size, void* d_ws, size_t ws_size,
                              hipStream_t stream)
{
    const float* enc = (const float*)d_in[0];
    const float* sem = (const float*)d_in[1];
    const float* wt[4] = { (const float*)d_in[2],  (const float*)d_in[6],
                           (const float*)d_in[10], (const float*)d_in[14] };
    const float* bt[4] = { (const float*)d_in[3],  (const float*)d_in[7],
                           (const float*)d_in[11], (const float*)d_in[15] };
    const float* wc[4] = { (const float*)d_in[4],  (const float*)d_in[8],
                           (const float*)d_in[12], (const float*)d_in[16] };
    const float* bc[4] = { (const float*)d_in[5],  (const float*)d_in[9],
                           (const float*)d_in[13], (const float*)d_in[17] };
    const float* w_attn = (const float*)d_in[18];
    const float* b_attn = (const float*)d_in[19];
    const float* w_fin  = (const float*)d_in[20];
    const float* b_fin  = (const float*)d_in[21];
    float* out = (float*)d_out;

    const int B = 8, HW = 256 * 256;

    // ---- workspace layout (f16 elements; every block 32B-aligned) ----
    _Float16* actA  = (_Float16*)d_ws;           // 16,777,216
    _Float16* actB  = actA + 16777216;           // 16,777,216
    _Float16* qrow  = actB + 16777216;           //  8,388,608  q [B,16,HW]
    _Float16* kbuf  = qrow + 8388608;            //  8,388,608  kf [B,16,HW]
    _Float16* qtile = kbuf + 8388608;            //  8,388,608  q [B,HW,16]
    _Float16* encP  = qtile + 8388608;           //  1,048,576
    _Float16* semP  = encP + 1048576;            // 16,777,216
    _Float16* wp    = semP + 16777216;
    _Float16* wtA[4], *wcA[4];
    const int cinT[4]  = {512, 256, 128, 64};
    const int coutT[4] = {256, 128, 64, 32};
    for (int i = 0; i < 4; ++i) {
        wtA[i] = wp; wp += (size_t)cinT[i] * coutT[i] * 9;
        wcA[i] = wp; wp += (size_t)coutT[i] * coutT[i] * 9;
    }
    _Float16* wattnA = wp; wp += 512;
    float* attn = (float*)wp;                    // 2,048 f32

    // ---- pack weights + activations into WMMA-native layouts ----
    for (int i = 0; i < 4; ++i) {
        int tot = cinT[i] * coutT[i] * 9;
        pack_w_k<<<(tot + 255) / 256, 256, 0, stream>>>(wt[i], wtA[i],
                  cinT[i], coutT[i], 9, 1, tot);
        tot = coutT[i] * coutT[i] * 9;
        pack_w_k<<<(tot + 255) / 256, 256, 0, stream>>>(wc[i], wcA[i],
                  coutT[i], coutT[i], 9, 0, tot);
    }
    pack_w_k<<<2, 256, 0, stream>>>(w_attn, wattnA, 32, 16, 1, 0, 512);
    pack_act_k<<<(1048576 + 255) / 256, 256, 0, stream>>>(enc, encP, 256, 1048576);
    pack_act_k<<<(16777216 + 255) / 256, 256, 0, stream>>>(sem, semP, HW, 16777216);

    dim3 w32(32);
    // ---- decoder: (convT + ELU, conv3x3 + ELU) x 4 ----
    convT_elu_k<512,256><<<B * 16 * 32 * 2,  w32, 0, stream>>>(encP, wtA[0], bt[0], actA, 16, 16);
    conv3x3_tdm_k<256,256,32><<<B * 16 * 32 * 1,  64, 0, stream>>>(actA, wcA[0], bc[0], actB, 32, 32);
    convT_elu_k<256,128><<<B * 8 * 64 * 4,   w32, 0, stream>>>(actB, wtA[1], bt[1], actA, 32, 32);
    conv3x3_tdm_k<128,128,64><<<B * 8 * 64 * 1,  128, 0, stream>>>(actA, wcA[1], bc[1], actB, 64, 64);
    convT_elu_k<128,64><<<B * 4 * 128 * 8,   w32, 0, stream>>>(actB, wtA[2], bt[2], actA, 64, 64);
    conv3x3_tdm_k<64,64,64><<<B * 4 * 128 * 2,   128, 0, stream>>>(actA, wcA[2], bc[2], actB, 128, 128);
    convT_elu_k<64,32><<<B * 2 * 256 * 16,   w32, 0, stream>>>(actB, wtA[3], bt[3], actA, 128, 128);
    conv3x3_tdm_k<32,32,64><<<B * 2 * 256 * 4,   128, 0, stream>>>(actA, wcA[3], bc[3], actB, 256, 256);

    // ---- attention head ----
    conv1x1_k<8><<<B * (HW / 128), w32, 0, stream>>>(actB, wattnA, b_attn, qrow, qtile, HW);
    conv1x1_k<8><<<B * (HW / 128), w32, 0, stream>>>(semP, wattnA, b_attn, kbuf,
                                                     (_Float16*)nullptr, HW);
    zero_f32_k<<<(B * 256 + 255) / 256, 256, 0, stream>>>(attn, B * 256);
    attn_accum_k<<<B * 64, w32, 0, stream>>>(qrow, kbuf, attn, HW, 64);
    softmax16_k<<<1, 128, 0, stream>>>(attn, B * 16);
    fused_final_k<<<B * (HW / 16), w32, 0, stream>>>(qtile, attn, w_fin, b_fin, out, HW);
}
